// BahdanauAttention_31920196944495
// MI455X (gfx1250) — matrix operand order repaired and checked by reading, not compiled, not run on hardware
//
#include <hip/hip_runtime.h>
#include <hip/hip_bf16.h>
#include <math.h>

#define B_ 64
#define L_ 4096
#define D_ 512
#define H_ 512
#define U_ 256

typedef __attribute__((ext_vector_type(16))) __bf16 v16bf;
typedef __attribute__((ext_vector_type(8)))  float  v8f;
typedef __attribute__((ext_vector_type(4)))  float  v4f;

// ---------- kernel 1: W1^T -> bf16 [U, D] ----------
__global__ __launch_bounds__(256) void k_w1t(const float* __restrict__ W1k,
                                             __bf16* __restrict__ W1T) {
  int i = blockIdx.x * 256 + threadIdx.x;     // U_*D_ elements
  int u = i >> 9, d = i & (D_ - 1);
  W1T[i] = (__bf16)W1k[d * U_ + u];
}

// ---------- kernel 2: hbias[b,u] = hidden@W2 + W2_b + W1_b ----------
__global__ __launch_bounds__(256) void k_hbias(const float* __restrict__ hidden,
                                               const float* __restrict__ W2k,
                                               const float* __restrict__ W2b,
                                               const float* __restrict__ W1b,
                                               float* __restrict__ hbias) {
  int i = blockIdx.x * 256 + threadIdx.x;     // B_*U_ elements
  int b = i >> 8, u = i & (U_ - 1);
  const float* h = hidden + b * H_;
  float s = W2b[u] + W1b[u];
  for (int k = 0; k < H_; ++k) s = fmaf(h[k], W2k[k * U_ + u], s);
  hbias[i] = s;
}

// ---------- kernel 3: fused WMMA score GEMM + tanh + V-dot + exp + partial ctx ----------
// grid: 2048 blocks x 256 threads; block = 128 rows. Single HBM pass over features:
// the block's weighted-sum re-read hits L0/L2 (data just loaded by the GEMM).
// Softmax shift uses the fixed bound C = sum|V_k| (tanh in [-1,1]) -> shift-invariant
// softmax needs no global max; exp(s - C) in [e^-16, 1], numerically safe.
__global__ __launch_bounds__(256) void k_scores_fused(const float* __restrict__ feat,
                                                      const __bf16* __restrict__ W1T,
                                                      const float* __restrict__ hbias,
                                                      const float* __restrict__ Vk,
                                                      float* __restrict__ e_out,
                                                      float* __restrict__ ctx_part,
                                                      float* __restrict__ esum_part) {
  const int b     = blockIdx.x >> 5;                          // 32 chunks / batch
  const int chunk = blockIdx.x & 31;
  const int wid   = threadIdx.x >> 5;
  const int lr    = chunk * 128 + wid * 16;
  const int lane  = threadIdx.x & 31;
  const int mrow  = lane & 15;
  const int khalf = lane >> 4;

  __shared__ float esh[128];
  __shared__ float red[128];

  // A fragment source: row (lr+mrow), K base offset per lane half
  const float* arow = feat + ((size_t)b * L_ + lr + mrow) * D_ + khalf * 16;

  v8f acc[16];
#pragma unroll
  for (int n = 0; n < 16; ++n) acc[n] = (v8f)(0.f);

#pragma unroll 1
  for (int k = 0; k < D_ / 32; ++k) {
    const float* ap = arow + k * 32;
    v4f f0 = *(const v4f*)(ap);
    v4f f1 = *(const v4f*)(ap + 4);
    v4f f2 = *(const v4f*)(ap + 8);
    v4f f3 = *(const v4f*)(ap + 12);
    __builtin_prefetch(ap + 32, 0, 1);        // next K-step of features

    v16bf a;
#pragma unroll
    for (int i = 0; i < 4; ++i) {
      a[i]      = (__bf16)f0[i];
      a[4 + i]  = (__bf16)f1[i];
      a[8 + i]  = (__bf16)f2[i];
      a[12 + i] = (__bf16)f3[i];
    }

    // B fragment: column N = n*16 + mrow of W1 == row of W1T, 16 contiguous K
    const __bf16* bp = W1T + (size_t)mrow * D_ + k * 32 + khalf * 16;
#pragma unroll
    for (int n = 0; n < 16; ++n) {
      v16bf bb = *(const v16bf*)(bp + (size_t)n * 16 * D_);
      acc[n] = __builtin_amdgcn_wmma_f32_16x16x32_bf16(
          /*neg_a=*/false, a, /*neg_b=*/false, bb,
          /*c_mod=*/(short)0, acc[n], /*reuse_a=*/false, /*reuse_b=*/false);
    }
  }

  // ---- epilogue: score -> e = exp(score - sum|V|)  (V_b shift cancels) ----
  float hb[16], vv[16];
#pragma unroll
  for (int n = 0; n < 16; ++n) {
    hb[n] = hbias[b * U_ + n * 16 + mrow];
    vv[n] = Vk[n * 16 + mrow];
  }
  float sav = 0.f;                              // sum|V_k| (softmax shift bound)
#pragma unroll
  for (int n = 0; n < 16; ++n) sav += fabsf(vv[n]);
#pragma unroll
  for (int off = 1; off < 16; off <<= 1) sav += __shfl_xor(sav, off, 32);

  float* erow = e_out + (size_t)b * L_ + lr;
#pragma unroll
  for (int j = 0; j < 8; ++j) {
    float s = 0.f;
#pragma unroll
    for (int n = 0; n < 16; ++n) s += tanhf(acc[n][j] + hb[n]) * vv[n];
    // reduce across the 16-lane half (C layout: lanes0-15 -> M=j, lanes16-31 -> M=j+8)
#pragma unroll
    for (int off = 1; off < 16; off <<= 1) s += __shfl_xor(s, off, 32);
    const float e = __expf(s - sav);
    if (lane == 0)  { erow[j]     = e; esh[wid * 16 + j]     = e; }
    if (lane == 16) { erow[j + 8] = e; esh[wid * 16 + 8 + j] = e; }
  }
  __syncthreads();

  // ---- block partial sum of e (deterministic tree) ----
  const int t = threadIdx.x;
  if (t < 128) red[t] = esh[t];
  __syncthreads();
  for (int s = 64; s > 0; s >>= 1) { if (t < s) red[t] += red[t + s]; __syncthreads(); }
  if (t == 0) esum_part[b * 32 + chunk] = red[0];

  // ---- block partial context: sum_l e_l * features[l, :]  (L0/L2-resident re-read) ----
  const float* fbase = feat + ((size_t)b * L_ + chunk * 128) * D_;
  float a0 = 0.f, a1 = 0.f;
  for (int l = 0; l < 128; ++l) {
    const float ww = esh[l];
    a0 = fmaf(ww, fbase[(size_t)l * D_ + t],       a0);
    a1 = fmaf(ww, fbase[(size_t)l * D_ + t + 256], a1);
  }
  float* pp = ctx_part + ((size_t)chunk * B_ + b) * D_;
  pp[t] = a0; pp[t + 256] = a1;
}

// ---------- kernel 4: per-batch 1/sum(e) ----------
__global__ __launch_bounds__(64) void k_esum(const float* __restrict__ esum_part,
                                             float* __restrict__ inv_total) {
  const int b = threadIdx.x;                  // 64 threads, 1 block
  float s = 0.f;
#pragma unroll
  for (int c = 0; c < 32; ++c) s += esum_part[b * 32 + c];
  inv_total[b] = 1.f / s;
}

// ---------- kernel 5: reduce ctx partials, normalize ----------
__global__ __launch_bounds__(256) void k_finalize_ctx(const float* __restrict__ ctx_part,
                                                      const float* __restrict__ inv_total,
                                                      float* __restrict__ ctx) {
  int i = blockIdx.x * 256 + threadIdx.x;     // B_*D_ elements
  int b = i >> 9;
  float s = 0.f;
#pragma unroll
  for (int c = 0; c < 32; ++c) s += ctx_part[((size_t)c * B_ + (size_t)b) * D_ + (i & (D_ - 1))];
  ctx[i] = s * inv_total[b];
}

// ---------- kernel 6: normalize stored e into attention weights ----------
__global__ __launch_bounds__(256) void k_scale_w(float* __restrict__ w,
                                                 const float* __restrict__ inv_total) {
  int i = blockIdx.x * 256 + threadIdx.x;     // B_*L_ elements
  w[i] = w[i] * inv_total[i >> 12];
}

extern "C" void kernel_launch(void* const* d_in, const int* in_sizes, int n_in,
                              void* d_out, int out_size, void* d_ws, size_t ws_size,
                              hipStream_t stream) {
  const float* features = (const float*)d_in[0];
  const float* hidden   = (const float*)d_in[1];
  const float* W1k      = (const float*)d_in[2];
  const float* W1b      = (const float*)d_in[3];
  const float* W2k      = (const float*)d_in[4];
  const float* W2b      = (const float*)d_in[5];
  const float* Vk       = (const float*)d_in[6];
  // V_b (d_in[7]) cancels in the shifted softmax; scores are only used via softmax.

  float* ctx = (float*)d_out;                 // [B, D] = 32768 floats
  float* wts = (float*)d_out + B_ * D_;       // [B, L] = 262144 floats

  char* ws = (char*)d_ws;
  __bf16* W1T  = (__bf16*)ws;                                   // 256 KB @ 0
  float*  hb   = (float*)(ws + 262144);                         // 64 KB
  float*  esum = (float*)(ws + 327680);                         // 8 KB  [B][32]
  float*  invt = (float*)(ws + 335872);                         // 256 B [B]
  float*  part = (float*)(ws + 336128);                         // 4 MB  [32][B][D]

  k_w1t   <<<(U_ * D_) / 256, 256, 0, stream>>>(W1k, W1T);
  k_hbias <<<(B_ * U_) / 256, 256, 0, stream>>>(hidden, W2k, W2b, W1b, hb);
  k_scores_fused<<<(B_ * L_) / 128, 256, 0, stream>>>(features, W1T, hb, Vk,
                                                      wts, part, esum);
  k_esum        <<<1, 64, 0, stream>>>(esum, invt);
  k_finalize_ctx<<<(B_ * D_) / 256, 256, 0, stream>>>(part, invt, ctx);
  k_scale_w     <<<(B_ * L_) / 256, 256, 0, stream>>>(wts, invt);
}
